// LinearAttention_4681514353452
// MI455X (gfx1250) — compile-verified, matrix operand order checked by reading, not run on hardware
//
#include <hip/hip_runtime.h>

// ---------------------------------------------------------------------------
// LinearAttention for MI455X (gfx1250): bf16 WMMA GEMMs + WMMA kv/num,
// async-to-LDS staging for K/V tiles.
// B=4, T=4096, C=1024, H=16, D=64.
// ---------------------------------------------------------------------------

constexpr int Bc = 4;
constexpr int Tc = 4096;
constexpr int Cc = 1024;
constexpr int Hh = 16;
constexpr int Dd = 64;

typedef __bf16  bf16x16 __attribute__((ext_vector_type(16)));
typedef float   f32x8   __attribute__((ext_vector_type(8)));

union Frag {
    bf16x16        v;
    uint4          u[2];
    unsigned short s[16];
};

__device__ __forceinline__ float bf2f(unsigned short u) {
    union { unsigned int i; float f; } x;
    x.i = ((unsigned int)u) << 16;
    return x.f;
}

__device__ __forceinline__ unsigned short f2bf(float f) {
    union { float f; unsigned int u; } x;
    x.f = f;
    unsigned int r = x.u + 0x7FFFu + ((x.u >> 16) & 1u);   // round-to-nearest-even
    return (unsigned short)(r >> 16);
}

// Load a 16x32 bf16 fragment (A-style layout, lane = row, K packed along regs).
__device__ __forceinline__ bf16x16 load_frag(const unsigned short* __restrict__ base,
                                             int ld, int row0, int k0, int lane) {
    int r  = lane & 15;
    int kb = (lane >> 4) << 3;       // 0 or 8
    const unsigned short* p = base + (size_t)(row0 + r) * ld + k0 + kb;
    Frag f;
    f.u[0] = *reinterpret_cast<const uint4*>(p);        // K = k0+kb   .. +7
    f.u[1] = *reinterpret_cast<const uint4*>(p + 16);   // K = k0+16+kb.. +7
    return f.v;
}

__device__ __forceinline__ f32x8 wmma_bf16(bf16x16 a, bf16x16 b, f32x8 c) {
    return __builtin_amdgcn_wmma_f32_16x16x32_bf16(
        /*neg_a=*/false, a, /*neg_b=*/false, b,
        /*c_mod=*/(short)0, c, /*reuse_a=*/false, /*reuse_b=*/false);
}

// Async memory->LDS 16-byte copy (CDNA5 GLOBAL_LOAD_ASYNC_TO_LDS_B128).
// lds_ptr: generic pointer to __shared__ (low 32 bits = LDS byte offset).
__device__ __forceinline__ void async_ld_b128(void* lds_ptr, const void* gsrc) {
    unsigned           ldst = (unsigned)(size_t)lds_ptr;
    unsigned long long ga   = (unsigned long long)(size_t)gsrc;
    asm volatile("global_load_async_to_lds_b128 %0, %1, off"
                 :: "v"(ldst), "v"(ga) : "memory");
}
__device__ __forceinline__ void wait_async0() {
    asm volatile("s_wait_asynccnt 0x0" ::: "memory");
}

// ---------------------------------------------------------------------------
// f32 -> bf16 conversion, 8 elements per thread (b128 in/out)
// ---------------------------------------------------------------------------
__global__ void __launch_bounds__(256)
f2bf8_kernel(const float4* __restrict__ src, uint4* __restrict__ dst, size_t n8) {
    size_t i = (size_t)blockIdx.x * 256 + threadIdx.x;
    if (i < n8) {
        float4 a = src[2 * i], b = src[2 * i + 1];
        union { unsigned short s[8]; uint4 u; } o;
        o.s[0] = f2bf(a.x); o.s[1] = f2bf(a.y); o.s[2] = f2bf(a.z); o.s[3] = f2bf(a.w);
        o.s[4] = f2bf(b.x); o.s[5] = f2bf(b.y); o.s[6] = f2bf(b.z); o.s[7] = f2bf(b.w);
        dst[i] = o.u;
    }
}

__global__ void __launch_bounds__(256)
zero_kernel(float* __restrict__ p, size_t n) {
    size_t i = (size_t)blockIdx.x * 256 + threadIdx.x;
    if (i < n) p[i] = 0.0f;
}

// ---------------------------------------------------------------------------
// GEMM: out[m,n] = sum_k A[m,k] * W[n,k] + bias[n]        (y = A @ W^T + b)
// mode 0: store bf16;  mode 1: ELU+1 then bf16;  mode 2: store f32.
// Block: 256 threads = 8 waves (2x4), block tile 128(M) x 256(N), wave 64x64.
// ---------------------------------------------------------------------------
__global__ void __launch_bounds__(256)
gemm_bf16_kernel(const unsigned short* __restrict__ A,
                 const unsigned short* __restrict__ W,
                 const float* __restrict__ bias,
                 unsigned short* __restrict__ outB,
                 float* __restrict__ outF,
                 int M, int N, int K, int mode) {
    const int tid  = threadIdx.x;
    const int lane = tid & 31;
    const int w    = tid >> 5;
    const int m0   = blockIdx.y * 128 + (w >> 2) * 64;
    const int n0   = blockIdx.x * 256 + (w & 3) * 64;
    const int rlo  = lane & 15;
    const int rhi  = (lane >> 4) << 3;

    f32x8 acc[4][4] = {};
    for (int k0 = 0; k0 < K; k0 += 32) {
        bf16x16 a[4], b[4];
#pragma unroll
        for (int i = 0; i < 4; ++i) a[i] = load_frag(A, K, m0 + 16 * i, k0, lane);
#pragma unroll
        for (int j = 0; j < 4; ++j) b[j] = load_frag(W, K, n0 + 16 * j, k0, lane);
#pragma unroll
        for (int i = 0; i < 4; ++i)
#pragma unroll
            for (int j = 0; j < 4; ++j)
                acc[i][j] = wmma_bf16(a[i], b[j], acc[i][j]);
    }

    float bj[4];
#pragma unroll
    for (int j = 0; j < 4; ++j) bj[j] = bias[n0 + 16 * j + rlo];

#pragma unroll
    for (int i = 0; i < 4; ++i)
#pragma unroll
        for (int j = 0; j < 4; ++j)
#pragma unroll
            for (int r = 0; r < 8; ++r) {
                int   m = m0 + 16 * i + r + rhi;
                int   n = n0 + 16 * j + rlo;
                float v = acc[i][j][r] + bj[j];
                if (mode == 1) {                          // elu(v)+1, branchless
                    float e = __expf(fminf(v, 0.0f));
                    v = (v > 0.0f) ? (v + 1.0f) : e;
                }
                size_t idx = (size_t)m * N + n;
                if (mode == 2) outF[idx] = v;
                else           outB[idx] = f2bf(v);
            }
}

// ---------------------------------------------------------------------------
// kv[b,h,d,e] += sum_{t in chunk} phiK[b,t,h,d] * V[b,t,h,e]   (K^T V, WMMA)
// ksum[b,h,d] += sum_{t in chunk} phiK[b,t,h,d]                (folded in)
// Grid: (B*H*8) blocks; block = one (b,h) x 512-t chunk; partials reduced via
// global_atomic_add_f32 into zero-initialized kv / ksum.
// K/V tiles staged memory->LDS with GLOBAL_LOAD_ASYNC_TO_LDS_B128.
// ---------------------------------------------------------------------------
__global__ void __launch_bounds__(256)
kv_kernel(const unsigned short* __restrict__ Kb,
          const unsigned short* __restrict__ Vb,
          float* __restrict__ KV, float* __restrict__ KS) {
    const int bh  = blockIdx.x >> 3;
    const int tch = blockIdx.x & 7;
    const int b   = bh >> 4;
    const int h   = bh & 15;
    const int tid = threadIdx.x, lane = tid & 31, w = tid >> 5;

    __shared__ __align__(16) unsigned short sk[32 * 64];
    __shared__ __align__(16) unsigned short sv[32 * 64];
    __shared__ float part[256];

    const int iT  = w & 3;             // d-tile (M)
    const int j0  = (w >> 2) * 2;      // e-tiles j0, j0+1 (N)
    const int row = tid >> 3;          // staging row (0..31)
    const int u   = tid & 7;           // staging 16B chunk
    const int rlo = lane & 15;
    const int kbT = (lane >> 4) << 3;
    const int dks = tid & 63;          // ksum column
    const int rg  = tid >> 6;          // ksum row group (8 rows each)

    f32x8 acc[2] = {};
    float ks_acc = 0.0f;

    for (int t0 = tch * 512; t0 < tch * 512 + 512; t0 += 32) {
        size_t g = ((size_t)(b * Tc + t0 + row)) * Cc + h * Dd;
        async_ld_b128(&sk[row * 64 + u * 8], Kb + g + u * 8);
        async_ld_b128(&sv[row * 64 + u * 8], Vb + g + u * 8);
        wait_async0();
        __syncthreads();

        // WMMA fragments (transposed access: contraction dim t is the LDS row)
        Frag fa;
#pragma unroll
        for (int e = 0; e < 8; ++e) {
            fa.s[e]     = sk[(kbT + e) * 64 + 16 * iT + rlo];
            fa.s[8 + e] = sk[(16 + kbT + e) * 64 + 16 * iT + rlo];
        }
#pragma unroll
        for (int jj = 0; jj < 2; ++jj) {
            Frag fb;
#pragma unroll
            for (int e = 0; e < 8; ++e) {
                fb.s[e]     = sv[(kbT + e) * 64 + 16 * (j0 + jj) + rlo];
                fb.s[8 + e] = sv[(16 + kbT + e) * 64 + 16 * (j0 + jj) + rlo];
            }
            acc[jj] = wmma_bf16(fa.v, fb.v, acc[jj]);
        }

        // ksum partial from the same staged K tile
#pragma unroll
        for (int r = 0; r < 8; ++r) ks_acc += bf2f(sk[(rg * 8 + r) * 64 + dks]);

        __syncthreads();
    }

    // kv partial -> global atomic
#pragma unroll
    for (int jj = 0; jj < 2; ++jj)
#pragma unroll
        for (int r = 0; r < 8; ++r) {
            int d = 16 * iT + r + ((lane >> 4) << 3);
            int e = 16 * (j0 + jj) + rlo;
            atomicAdd(&KV[(size_t)bh * 4096 + (size_t)d * 64 + e], acc[jj][r]);
        }

    // ksum partial -> global atomic
    part[tid] = ks_acc;
    __syncthreads();
    if (rg == 0)
        atomicAdd(&KS[(size_t)bh * 64 + dks],
                  part[dks] + part[64 + dks] + part[128 + dks] + part[192 + dks]);
}

// ---------------------------------------------------------------------------
// denom[b,t,h] = sum_d phiQ[b,t,h,d] * ksum[b,h,d]
// ---------------------------------------------------------------------------
__global__ void __launch_bounds__(256)
denom_kernel(const unsigned short* __restrict__ Qb, const float* __restrict__ KS,
             float* __restrict__ DEN) {
    size_t idx = (size_t)blockIdx.x * 256 + threadIdx.x;   // (b*T + t)*H + h
    int b   = (int)(idx / ((size_t)Tc * Hh));
    int rem = (int)(idx % ((size_t)Tc * Hh));
    int t   = rem / Hh;
    int h   = rem % Hh;
    const uint4* q4 = reinterpret_cast<const uint4*>(Qb + ((size_t)(b * Tc + t)) * Cc + h * Dd);
    const float* ks = KS + ((size_t)(b * Hh + h)) * 64;
    float s = 0.0f;
#pragma unroll
    for (int c = 0; c < 8; ++c) {
        union { uint4 u; unsigned short s[8]; } wq;
        wq.u = q4[c];
#pragma unroll
        for (int e = 0; e < 8; ++e) s += bf2f(wq.s[e]) * ks[c * 8 + e];
    }
    DEN[idx] = s;
}

// ---------------------------------------------------------------------------
// y[b,t,h*64+e] = (sum_d phiQ[b,t,h,d] * kv[b,h,d,e]) / max(denom, 1e-6)
// One block per (b,h, t-chunk of 512). kv -> bf16 LDS once, frags reused.
// ---------------------------------------------------------------------------
__global__ void __launch_bounds__(256)
num_kernel(const unsigned short* __restrict__ Qb, const float* __restrict__ KV,
           const float* __restrict__ DEN, unsigned short* __restrict__ Yb) {
    const int bh  = blockIdx.x >> 3;
    const int tch = blockIdx.x & 7;
    const int b = bh >> 4, h = bh & 15;
    const int tid = threadIdx.x, lane = tid & 31, w = tid >> 5;

    __shared__ unsigned short skv[64 * 64];   // kv in bf16, [d*64 + e]
    for (int idx = tid; idx < 4096; idx += 256)
        skv[idx] = f2bf(KV[(size_t)bh * 4096 + idx]);
    __syncthreads();

    const int rlo = lane & 15;
    const int kbD = (lane >> 4) << 3;

    bf16x16 bfr[2][4];
#pragma unroll
    for (int ks = 0; ks < 2; ++ks)
#pragma unroll
        for (int j = 0; j < 4; ++j) {
            Frag f;
#pragma unroll
            for (int e = 0; e < 8; ++e) {
                f.s[e]     = skv[(32 * ks + kbD + e) * 64 + 16 * j + rlo];
                f.s[8 + e] = skv[(32 * ks + 16 + kbD + e) * 64 + 16 * j + rlo];
            }
            bfr[ks][j] = f.v;
        }

    const unsigned short* qbase = Qb + (size_t)b * Tc * Cc + h * Dd;  // ld = Cc
    const int t0w = tch * 512 + w * 64;

    f32x8 acc[4][4] = {};
#pragma unroll
    for (int ks = 0; ks < 2; ++ks) {
        bf16x16 a[4];
#pragma unroll
        for (int i = 0; i < 4; ++i) a[i] = load_frag(qbase, Cc, t0w + 16 * i, 32 * ks, lane);
#pragma unroll
        for (int i = 0; i < 4; ++i)
#pragma unroll
            for (int j = 0; j < 4; ++j)
                acc[i][j] = wmma_bf16(a[i], bfr[ks][j], acc[i][j]);
    }

#pragma unroll
    for (int i = 0; i < 4; ++i)
#pragma unroll
        for (int j = 0; j < 4; ++j)
#pragma unroll
            for (int r = 0; r < 8; ++r) {
                int   t   = t0w + 16 * i + r + ((lane >> 4) << 3);
                int   e   = 16 * j + rlo;
                float den = DEN[((size_t)(b * Tc + t)) * Hh + h];
                den = fmaxf(den, 1e-6f);
                Yb[((size_t)(b * Tc + t)) * Cc + h * Dd + e] = f2bf(acc[i][j][r] / den);
            }
}

// ---------------------------------------------------------------------------
// Host launcher
// ---------------------------------------------------------------------------
extern "C" void kernel_launch(void* const* d_in, const int* in_sizes, int n_in,
                              void* d_out, int out_size, void* d_ws, size_t ws_size,
                              hipStream_t stream) {
    (void)in_sizes; (void)n_in; (void)out_size; (void)ws_size;

    const float* x  = (const float*)d_in[0];
    const float* Wq = (const float*)d_in[1];
    const float* bq = (const float*)d_in[2];
    const float* Wk = (const float*)d_in[3];
    const float* bk = (const float*)d_in[4];
    const float* Wv = (const float*)d_in[5];
    const float* bv = (const float*)d_in[6];
    const float* Wo = (const float*)d_in[7];
    const float* bo = (const float*)d_in[8];
    float* out = (float*)d_out;

    const size_t nX = (size_t)Bc * Tc * Cc;   // 16,777,216
    const size_t nW = (size_t)Cc * Cc;        //  1,048,576

    char*  ws  = (char*)d_ws;
    size_t off = 0;
    auto alloc = [&](size_t bytes) -> void* {
        void* p = ws + off;
        off = (off + bytes + 255) & ~(size_t)255;
        return p;
    };

    unsigned short* xb  = (unsigned short*)alloc(nX * 2);
    unsigned short* wqb = (unsigned short*)alloc(nW * 2);
    unsigned short* wkb = (unsigned short*)alloc(nW * 2);
    unsigned short* wvb = (unsigned short*)alloc(nW * 2);
    unsigned short* wob = (unsigned short*)alloc(nW * 2);
    unsigned short* qb  = (unsigned short*)alloc(nX * 2);
    unsigned short* kb  = (unsigned short*)alloc(nX * 2);
    unsigned short* vb  = (unsigned short*)alloc(nX * 2);
    float* kv   = (float*)alloc((size_t)Bc * Hh * Dd * Dd * 4);
    float* ksum = (float*)alloc((size_t)Bc * Hh * Dd * 4);
    float* den  = (float*)alloc((size_t)Bc * Tc * Hh * 4);
    unsigned short* yb = xb;   // x is dead after the QKV projections; reuse.

    // 1) conversions to bf16 (8 elems/thread)
    f2bf8_kernel<<<(unsigned)(nX / 8 / 256), 256, 0, stream>>>((const float4*)x,  (uint4*)xb,  nX / 8);
    f2bf8_kernel<<<(unsigned)(nW / 8 / 256), 256, 0, stream>>>((const float4*)Wq, (uint4*)wqb, nW / 8);
    f2bf8_kernel<<<(unsigned)(nW / 8 / 256), 256, 0, stream>>>((const float4*)Wk, (uint4*)wkb, nW / 8);
    f2bf8_kernel<<<(unsigned)(nW / 8 / 256), 256, 0, stream>>>((const float4*)Wv, (uint4*)wvb, nW / 8);
    f2bf8_kernel<<<(unsigned)(nW / 8 / 256), 256, 0, stream>>>((const float4*)Wo, (uint4*)wob, nW / 8);

    // zero kv + ksum accumulators
    {
        size_t nkv = (size_t)Bc * Hh * Dd * Dd;
        size_t nks = (size_t)Bc * Hh * Dd;
        zero_kernel<<<(unsigned)((nkv + 255) / 256), 256, 0, stream>>>(kv, nkv);
        zero_kernel<<<(unsigned)((nks + 255) / 256), 256, 0, stream>>>(ksum, nks);
    }

    // 2) Q/K/V projections (M=16384, N=K=1024); ELU+1 fused for Q,K
    const int M = Bc * Tc, N = Cc, K = Cc;
    dim3 ggrid(N / 256, M / 128);
    gemm_bf16_kernel<<<ggrid, 256, 0, stream>>>(xb, wqb, bq, qb, nullptr, M, N, K, 1);
    gemm_bf16_kernel<<<ggrid, 256, 0, stream>>>(xb, wkb, bk, kb, nullptr, M, N, K, 1);
    gemm_bf16_kernel<<<ggrid, 256, 0, stream>>>(xb, wvb, bv, vb, nullptr, M, N, K, 0);

    // 3) kv aggregation + ksum (t-chunked, atomic reduction)
    kv_kernel<<<Bc * Hh * 8, 256, 0, stream>>>(kb, vb, kv, ksum);

    // 4) denom
    denom_kernel<<<(Bc * Tc * Hh) / 256, 256, 0, stream>>>(qb, ksum, den);

    // 5) numerator + divide -> y (bf16, aliases xb)
    num_kernel<<<Bc * Hh * (Tc / 512), 256, 0, stream>>>(qb, kv, den, yb);

    // 6) output projection -> f32 out
    gemm_bf16_kernel<<<ggrid, 256, 0, stream>>>(yb, wob, bo, nullptr, out, M, N, K, 2);
}
